// LSTMCell_52115133170171
// MI455X (gfx1250) — matrix operand order repaired and checked by reading, not compiled, not run on hardware
//
#include <hip/hip_runtime.h>
#include <stdint.h>

#define B_  16384
#define I_  64
#define H_  1024
#define O_  64
#define K1  128   // I + O

typedef __attribute__((ext_vector_type(16))) __bf16    v16bf;
typedef __attribute__((ext_vector_type(8)))  float     v8f;
typedef __attribute__((ext_vector_type(4)))  unsigned int u32x4;

union FragU { u32x4 q[2]; v16bf v; };

__device__ __forceinline__ v16bf ld_frag16(const uint16_t* __restrict__ p0,
                                           const uint16_t* __restrict__ p1) {
  FragU f;
  f.q[0] = *(const u32x4*)p0;
  f.q[1] = *(const u32x4*)p1;
  return f.v;
}

__device__ __forceinline__ uint16_t f2bf(float f) {
  uint32_t u = __builtin_bit_cast(uint32_t, f);
  return (uint16_t)((u + 0x7FFFu + ((u >> 16) & 1u)) >> 16);
}

__device__ __forceinline__ float fsig(float x) {
  // 1/(1+exp(-x)) via v_exp_f32 (base-2) + v_rcp_f32
  return __builtin_amdgcn_rcpf(1.0f + __builtin_amdgcn_exp2f(-1.4426950408889634f * x));
}
__device__ __forceinline__ float ftanh(float x) {
  // tanh(x) = 2*sigmoid(2x) - 1
  return 2.0f * __builtin_amdgcn_rcpf(1.0f + __builtin_amdgcn_exp2f(-2.8853900817779268f * x)) - 1.0f;
}

// ---------------- pack kernels (fp32 -> bf16, WMMA-friendly layouts) ----------------

// A = [x | ym1] row-major bf16 [B][128]; one dword (2 bf16) per thread
__global__ __launch_bounds__(256) void pack_a(const float* __restrict__ x,
                                              const float* __restrict__ ym1,
                                              uint16_t* __restrict__ Apack) {
  int tid = blockIdx.x * blockDim.x + threadIdx.x;     // B_*64 dwords
  if (tid >= B_ * 64) return;
  int row = tid >> 6, j = tid & 63;
  const float* src = (j < 32) ? (x + (size_t)row * I_ + 2 * j)
                              : (ym1 + (size_t)row * O_ + (2 * j - 64));
  uint32_t v = (uint32_t)f2bf(src[0]) | ((uint32_t)f2bf(src[1]) << 16);
  ((uint32_t*)Apack)[tid] = v;
}

// Wt[g][n][k] bf16, k in [0,64)=W_g[k][n], [64,128)=R_g[k-64][n]  (N-major => B-frags contiguous)
__global__ __launch_bounds__(256) void pack_w(const float* __restrict__ Wz, const float* __restrict__ Wi,
                                              const float* __restrict__ Wf, const float* __restrict__ Wo,
                                              const float* __restrict__ Rz, const float* __restrict__ Ri,
                                              const float* __restrict__ Rf, const float* __restrict__ Ro,
                                              uint16_t* __restrict__ Wt) {
  int tid = blockIdx.x * blockDim.x + threadIdx.x;     // 4*1024*64 dwords
  if (tid >= 4 * 1024 * 64) return;
  int g = tid >> 16;
  int rem = tid & 65535;
  int n = rem >> 6, j = rem & 63;
  const float* W = (g == 0) ? Wz : (g == 1) ? Wi : (g == 2) ? Wf : Wo;
  const float* R = (g == 0) ? Rz : (g == 1) ? Ri : (g == 2) ? Rf : Ro;
  int k = 2 * j;
  float v0 = (k     < 64) ? W[(size_t)k * H_ + n]       : R[(size_t)(k - 64) * H_ + n];
  float v1 = (k + 1 < 64) ? W[(size_t)(k + 1) * H_ + n] : R[(size_t)(k + 1 - 64) * H_ + n];
  ((uint32_t*)Wt)[tid] = (uint32_t)f2bf(v0) | ((uint32_t)f2bf(v1) << 16);
}

// Pt[o][k] bf16 = proj[k][o]
__global__ __launch_bounds__(256) void pack_p(const float* __restrict__ proj,
                                              uint16_t* __restrict__ Pt) {
  int tid = blockIdx.x * blockDim.x + threadIdx.x;     // 64*512 dwords
  if (tid >= 64 * 512) return;
  int n = tid >> 9, j = tid & 511;
  int k = 2 * j;
  float v0 = proj[(size_t)k * O_ + n];
  float v1 = proj[(size_t)(k + 1) * O_ + n];
  ((uint32_t*)Pt)[tid] = (uint32_t)f2bf(v0) | ((uint32_t)f2bf(v1) << 16);
}

// ---------------- fused LSTM cell kernel ----------------
// grid = B/16 blocks of 256 threads (8 wave32). Block: 16 rows x 1024 cols.
// Wave w: columns [w*128, w*128+128) -> 8 N-tiles of 16.

__global__ __launch_bounds__(256)
void lstm_main(const float* __restrict__ c_in,
               const float* __restrict__ pi, const float* __restrict__ pf,
               const float* __restrict__ po,
               const float* __restrict__ bz, const float* __restrict__ bi,
               const float* __restrict__ bfg, const float* __restrict__ bo,
               const uint16_t* __restrict__ Apack,
               const uint16_t* __restrict__ Wt,
               const uint16_t* __restrict__ Pt,
               float* __restrict__ c_out, float* __restrict__ y_out)
{
  __shared__ __align__(16) uint16_t h_lds[8 * 16 * 128];   // per-wave 16x128 bf16 h tile
  __shared__ __align__(16) float    y_lds[8 * 16 * 64];    // per-wave y partials

  const int tid   = threadIdx.x;
  const int wave  = tid >> 5;
  const int lane  = tid & 31;
  const int lhalf = lane >> 4;          // 0: lanes 0-15, 1: lanes 16-31
  const int l16   = lane & 15;
  const int m0    = blockIdx.x * 16;    // batch row base
  const int nb    = wave * 128;         // wave's H column base

  // ---- A fragments: rows m0..m0+15, K=128 (4 x 16x32 bf16 fragments) ----
  // 16-bit A layout: lanes 0-15 row M=lane, chunks k0+[0..7] / k0+[16..23];
  //                  lanes 16-31 row M=lane-16, chunks k0+[8..15] / k0+[24..31].
  v16bf afrag[4];
  {
    const uint16_t* arow = Apack + (size_t)(m0 + l16) * K1;
#pragma unroll
    for (int s = 0; s < 4; ++s) {
      int k0 = s * 32 + lhalf * 8;
      afrag[s] = ld_frag16(arow + k0, arow + k0 + 16);
    }
  }

  for (int t = 0; t < 8; ++t) {
    const int ncol = nb + t * 16 + l16;       // this lane's global H column (C-layout N)
    // B-frag: lane = column; lanes 0-15 K=0..7,16..23, lanes 16-31 K=8..15,24..31 in Wt
    auto gate_acc = [&](int g) -> v8f {
      v8f acc = {};
      const uint16_t* wc = Wt + ((size_t)g * H_ + ncol) * K1 + lhalf * 8;
#pragma unroll
      for (int s = 0; s < 4; ++s) {
        v16bf bfrag = ld_frag16(wc + s * 32, wc + s * 32 + 16);
        acc = __builtin_amdgcn_wmma_f32_16x16x32_bf16(false, afrag[s], false, bfrag,
                                                      (short)0, acc, false, false);
      }
      return acc;
    };

    const float pi_v = pi[ncol], pf_v = pf[ncol], po_v = po[ncol];
    const float bz_v = bz[ncol], bi_v = bi[ncol], bf_v = bfg[ncol], bo_v = bo[ncol];
    const int rowb = m0 + lhalf * 8;          // C-layout: VGPR r -> row rowb + r

    float ct[8];
#pragma unroll
    for (int r = 0; r < 8; ++r) ct[r] = c_in[(size_t)(rowb + r) * H_ + ncol];

    v8f az = gate_acc(0);
    v8f ai = gate_acc(1);
    float t1[8];
#pragma unroll
    for (int r = 0; r < 8; ++r)
      t1[r] = ftanh(az[r] + bz_v) * fsig(ai[r] + pi_v * ct[r] + bi_v);

    v8f af = gate_acc(2);
    float cn[8];
#pragma unroll
    for (int r = 0; r < 8; ++r)
      cn[r] = t1[r] + ct[r] * fsig(af[r] + pf_v * ct[r] + bf_v);

    v8f ao = gate_acc(3);
#pragma unroll
    for (int r = 0; r < 8; ++r) {
      float o = fsig(ao[r] + po_v * cn[r] + bo_v);
      float h = ftanh(cn[r]) * o;
      c_out[(size_t)(rowb + r) * H_ + ncol] = cn[r];
      h_lds[wave * 2048 + (lhalf * 8 + r) * 128 + t * 16 + l16] = f2bf(h);
    }
  }

  // ---- y partial: h[16 x 128] @ proj[128-slice x 64], K local = 128 -> 4 k-steps ----
  v8f accy[4] = {v8f{}, v8f{}, v8f{}, v8f{}};
#pragma unroll
  for (int s2 = 0; s2 < 4; ++s2) {
    FragU fh;
    const int abase = wave * 2048 + l16 * 128 + s2 * 32 + lhalf * 8;
    fh.q[0] = *(const u32x4*)&h_lds[abase];
    fh.q[1] = *(const u32x4*)&h_lds[abase + 16];
    const v16bf ah = fh.v;
    const int kg = nb + s2 * 32 + lhalf * 8;     // global H index for proj rows
#pragma unroll
    for (int nt = 0; nt < 4; ++nt) {
      const uint16_t* pc = Pt + (size_t)(nt * 16 + l16) * H_ + kg;
      v16bf bp = ld_frag16(pc, pc + 16);
      accy[nt] = __builtin_amdgcn_wmma_f32_16x16x32_bf16(false, ah, false, bp,
                                                         (short)0, accy[nt], false, false);
    }
  }

  // stash f32 partials (per-wave region), then fixed-order cross-wave reduction
#pragma unroll
  for (int nt = 0; nt < 4; ++nt)
#pragma unroll
    for (int r = 0; r < 8; ++r)
      y_lds[wave * 1024 + (lhalf * 8 + r) * 64 + nt * 16 + l16] = accy[nt][r];

  __syncthreads();

#pragma unroll
  for (int e4 = 0; e4 < 4; ++e4) {
    int e = tid * 4 + e4;                      // 16*64 = 1024 elems / 256 threads
    float s = 0.0f;
#pragma unroll
    for (int w = 0; w < 8; ++w) s += y_lds[w * 1024 + e];
    int m = e >> 6, o = e & 63;
    y_out[(size_t)(m0 + m) * O_ + o] = s;
  }
}

// ---------------- host launcher ----------------

extern "C" void kernel_launch(void* const* d_in, const int* in_sizes, int n_in,
                              void* d_out, int out_size, void* d_ws, size_t ws_size,
                              hipStream_t stream) {
  (void)in_sizes; (void)n_in; (void)out_size; (void)ws_size;
  const float* x    = (const float*)d_in[0];
  const float* c    = (const float*)d_in[1];
  const float* ym1  = (const float*)d_in[2];
  const float* Wz   = (const float*)d_in[3];
  const float* Wi   = (const float*)d_in[4];
  const float* Wf   = (const float*)d_in[5];
  const float* Wo   = (const float*)d_in[6];
  const float* Rz   = (const float*)d_in[7];
  const float* Ri   = (const float*)d_in[8];
  const float* Rf   = (const float*)d_in[9];
  const float* Ro   = (const float*)d_in[10];
  const float* pi   = (const float*)d_in[11];
  const float* pf   = (const float*)d_in[12];
  const float* po   = (const float*)d_in[13];
  const float* bz   = (const float*)d_in[14];
  const float* bi   = (const float*)d_in[15];
  const float* bfg  = (const float*)d_in[16];
  const float* bo   = (const float*)d_in[17];
  const float* proj = (const float*)d_in[18];

  uint16_t* Apack = (uint16_t*)d_ws;                       // [B][128] bf16    (4 MiB)
  uint16_t* Wt    = Apack + (size_t)B_ * K1;               // [4][1024][128]   (1 MiB)
  uint16_t* Pt    = Wt + (size_t)4 * H_ * K1;              // [64][1024]       (128 KiB)

  float* c_out = (float*)d_out;
  float* y_out = c_out + (size_t)B_ * H_;

  pack_a<<<(B_ * 64 + 255) / 256, 256, 0, stream>>>(x, ym1, Apack);
  pack_w<<<(4 * 1024 * 64 + 255) / 256, 256, 0, stream>>>(Wz, Wi, Wf, Wo, Rz, Ri, Rf, Ro, Wt);
  pack_p<<<(64 * 512 + 255) / 256, 256, 0, stream>>>(proj, Pt);
  lstm_main<<<B_ / 16, 256, 0, stream>>>(c, pi, pf, po, bz, bi, bfg, bo,
                                         Apack, Wt, Pt, c_out, y_out);
}
